// AgentScaleDotProductAttention_15891378995540
// MI455X (gfx1250) — compile-verified, matrix operand order checked by reading, not run on hardware
//
#include <hip/hip_runtime.h>
#include <hip/hip_bf16.h>

#define BATCH  4
#define SEQ    4096
#define DMODEL 1024
#define AGENTS 64
#define HEADS  16
#define HDIM   64
#define SCALE  0.125f   // 1/sqrt(64)

typedef __attribute__((ext_vector_type(16))) __bf16 v16bf;
typedef __attribute__((ext_vector_type(8)))  float  v8f;

union Frag {
    v16bf          bf;
    unsigned short u[16];
    unsigned       d[8];
    uint4          q[2];
};

#define V8F_ZERO {0.f,0.f,0.f,0.f,0.f,0.f,0.f,0.f}

// Pack two f32 into one packed-bf16 dword with a single v_perm_b32
// (round-half-up via +0x8000, then byte-select the high halves).
__device__ __forceinline__ unsigned pack_bf16(float f0, float f1) {
    union { float f; unsigned u; } a, b; a.f = f0; b.f = f1;
    return __builtin_amdgcn_perm(b.u + 0x8000u, a.u + 0x8000u, 0x07060302u);
}

__device__ __forceinline__ unsigned short f2bf(float f) {
    union { float f; unsigned u; } v; v.f = f;
    return (unsigned short)((v.u + 0x8000u) >> 16);
}

__device__ __forceinline__ v8f wmma_bf16(v16bf a, v16bf b, v8f c) {
    return __builtin_amdgcn_wmma_f32_16x16x32_bf16(false, a, false, b, (short)0, c, false, false);
}

// A/B fragment from f32 row-major [16 rows x K], K contiguous, converted to bf16.
// ISA 16-bit 16x32 layout: lane = row (mod 16), half = lane>>4,
// element e -> k = e + (e&8) + half*8  => two contiguous 8-float runs.
__device__ __forceinline__ v16bf frag_f32_k(const float* __restrict__ base, int ldk, int lane) {
    const int half = lane >> 4, r = lane & 15;
    const float* p = base + (size_t)r * ldk + half * 8;
    float4 a = *reinterpret_cast<const float4*>(p + 0);
    float4 b = *reinterpret_cast<const float4*>(p + 4);
    float4 c = *reinterpret_cast<const float4*>(p + 16);
    float4 d = *reinterpret_cast<const float4*>(p + 20);
    Frag fr;
    fr.d[0] = pack_bf16(a.x, a.y); fr.d[1] = pack_bf16(a.z, a.w);
    fr.d[2] = pack_bf16(b.x, b.y); fr.d[3] = pack_bf16(b.z, b.w);
    fr.d[4] = pack_bf16(c.x, c.y); fr.d[5] = pack_bf16(c.z, c.w);
    fr.d[6] = pack_bf16(d.x, d.y); fr.d[7] = pack_bf16(d.z, d.w);
    return fr.bf;
}

// Same layout but source already bf16 (row-major, K contiguous, 16B aligned rows).
__device__ __forceinline__ v16bf frag_bf16_k(const unsigned short* __restrict__ base, int ldk, int lane) {
    const int half = lane >> 4, r = lane & 15;
    const unsigned short* p = base + (size_t)r * ldk + half * 8;
    Frag fr;
    fr.q[0] = *reinterpret_cast<const uint4*>(p);       // k .. k+7
    fr.q[1] = *reinterpret_cast<const uint4*>(p + 16);  // k+16 .. k+23
    return fr.bf;
}

// ---------------- Kernel 1: agent_tokens = Q[:, :64] @ W^T + bias  (bf16 out) ----------------
__global__ __launch_bounds__(256)
void k_agent_tokens(const float* __restrict__ q, const float* __restrict__ w,
                    const float* __restrict__ bias, unsigned short* __restrict__ at) {
    const int lane = threadIdx.x & 31;
    const int wid  = (blockIdx.x * blockDim.x + threadIdx.x) >> 5;  // 1024 waves
    const int b  = wid >> 8;
    const int mt = (wid >> 6) & 3;   // agent tile (M=64)
    const int nt = wid & 63;         // d tile (N=1024)
    const int half = lane >> 4, r = lane & 15;

    const float* qb = q + ((size_t)b * SEQ + mt * 16) * DMODEL;
    const float* wb = w + (size_t)(nt * 16) * DMODEL;   // W[n,k], k contiguous (== W^T column)
    v8f acc = V8F_ZERO;
    for (int k0 = 0; k0 < DMODEL; k0 += 32) {
        v16bf af = frag_f32_k(qb + k0, DMODEL, lane);
        v16bf bf = frag_f32_k(wb + k0, DMODEL, lane);
        acc = wmma_bf16(af, bf, acc);
    }
    float bn = bias[nt * 16 + r];
    unsigned short* dst = at + ((size_t)b * AGENTS + mt * 16 + half * 8) * DMODEL + nt * 16 + r;
#pragma unroll
    for (int j = 0; j < 8; j++)
        dst[(size_t)j * DMODEL] = f2bf(acc[j] + bn);
}

// ---------------- Kernel 2a: stage-1 logits = scale * AT @ K^T  (f32 out) ----------------
__global__ __launch_bounds__(256)
void k_s1_logits(const unsigned short* __restrict__ at, const float* __restrict__ kk,
                 float* __restrict__ logits) {
    const int lane = threadIdx.x & 31;
    const int wid  = (blockIdx.x * blockDim.x + threadIdx.x) >> 5;  // 4096 waves
    const int b  = wid >> 10;
    const int mt = (wid >> 8) & 3;   // agent tile
    const int nt = wid & 255;        // s tile (N=4096)
    const int half = lane >> 4, r = lane & 15;

    const unsigned short* ab = at + ((size_t)b * AGENTS + mt * 16) * DMODEL;
    const float* kb = kk + ((size_t)b * SEQ + nt * 16) * DMODEL;    // K[s,k], k contiguous
    v8f acc = V8F_ZERO;
    for (int k0 = 0; k0 < DMODEL; k0 += 32) {
        __builtin_prefetch(kb + k0 + 32, 0, 0);
        v16bf af = frag_bf16_k(ab + k0, DMODEL, lane);
        v16bf bf = frag_f32_k(kb + k0, DMODEL, lane);
        acc = wmma_bf16(af, bf, acc);
    }
    float* dst = logits + ((size_t)b * AGENTS + mt * 16 + half * 8) * SEQ + nt * 16 + r;
#pragma unroll
    for (int j = 0; j < 8; j++)
        dst[(size_t)j * SEQ] = acc[j] * SCALE;
}

// ---------------- Kernel 2b: row softmax over s=4096, write a1 as bf16 ----------------
__global__ __launch_bounds__(256)
void k_s1_softmax(const float* __restrict__ logits, unsigned short* __restrict__ a1) {
    const int row = blockIdx.x;                       // 256 rows (b*agents)
    const float* src = logits + (size_t)row * SEQ;
    unsigned short* dst = a1 + (size_t)row * SEQ;
    __shared__ float red[256];
    const int t = threadIdx.x;

    float m = -1e30f;
    for (int i = t; i < SEQ; i += 256) m = fmaxf(m, src[i]);
    red[t] = m; __syncthreads();
    for (int s2 = 128; s2 > 0; s2 >>= 1) { if (t < s2) red[t] = fmaxf(red[t], red[t + s2]); __syncthreads(); }
    m = red[0]; __syncthreads();

    float sum = 0.f;
    for (int i = t; i < SEQ; i += 256) sum += __expf(src[i] - m);
    red[t] = sum; __syncthreads();
    for (int s2 = 128; s2 > 0; s2 >>= 1) { if (t < s2) red[t] += red[t + s2]; __syncthreads(); }
    float inv = 1.0f / red[0];

    for (int i = t; i < SEQ; i += 256) dst[i] = f2bf(__expf(src[i] - m) * inv);
}

// ---------------- Kernel 2c: agent_values^T = (a1 @ V)^T  (bf16, [b][d][agents]) ----------------
// 4 waves per block share one (batch, d-tile); V tile staged through LDS with
// global_load_async_to_lds_b128 (ASYNCcnt path), one b128 async copy per wave per K-step.
__global__ __launch_bounds__(128)
void k_s1_av(const unsigned short* __restrict__ a1, const float* __restrict__ v,
             unsigned short* __restrict__ avT) {
    __shared__ float vtile[32 * 16];                  // [k][n] f32, 2 KB
    const int lane = threadIdx.x & 31;
    const int wsub = threadIdx.x >> 5;                // 0..3 = agent tile (M=64)
    const int b  = blockIdx.x >> 6;                   // 256 blocks = (batch, d-tile)
    const int nt = blockIdx.x & 63;
    const int half = lane >> 4, r = lane & 15;

    const unsigned short* ab = a1 + ((size_t)b * AGENTS + wsub * 16) * SEQ;
    const float* vb = v + (size_t)b * SEQ * DMODEL + nt * 16;   // V[k=s][n=d], n contiguous

    // this wave async-copies rows k = wsub*8 .. wsub*8+7 of the 32x16 tile
    const int krow = wsub * 8 + (lane >> 2);
    const int ncol = (lane & 3) * 4;
    const unsigned ldsoff = (unsigned)(uintptr_t)&vtile[krow * 16 + ncol];  // LDS byte offset (low 32 bits)

    v8f acc = V8F_ZERO;
    for (int k0 = 0; k0 < SEQ; k0 += 32) {
        const float* gsrc = vb + (size_t)(k0 + krow) * DMODEL + ncol;
        asm volatile("global_load_async_to_lds_b128 %0, %1, off"
                     :: "v"(ldsoff), "v"(gsrc) : "memory");
        asm volatile("s_wait_asynccnt 0x0" ::: "memory");
        __syncthreads();
        // B fragment from LDS f32 [k][n]: element e -> k = e + (e&8) + half*8, n = r
        Frag fr;
#pragma unroll
        for (int e = 0; e < 16; e += 2) {
            int ka = e + (e & 8) + half * 8;     // e, e+1 map to adjacent k
            fr.d[e >> 1] = pack_bf16(vtile[ka * 16 + r], vtile[(ka + 1) * 16 + r]);
        }
        v16bf af = frag_bf16_k(ab + k0, SEQ, lane);
        acc = wmma_bf16(af, fr.bf, acc);
        __syncthreads();                         // tile consumed; safe to overwrite
    }
    // avT[b][n][m], m contiguous: stage-3 B operand becomes K-contiguous rows
    unsigned short* dst = avT + ((size_t)b * DMODEL + nt * 16 + r) * AGENTS + wsub * 16 + half * 8;
#pragma unroll
    for (int j = 0; j < 8; j++)
        dst[j] = f2bf(acc[j]);
}

// ---------------- Kernel 3: fused stage-2 (logits -> softmax(64) -> out, attention) ----------------
__global__ __launch_bounds__(256)
void k_s2_fused(const float* __restrict__ q, const unsigned short* __restrict__ at,
                const unsigned short* __restrict__ avT,
                float* __restrict__ out, float* __restrict__ attn) {
    __shared__ unsigned short lds_a2[8 * 16 * 64];   // per-wave 16x64 bf16 a2 tile (16 KB)
    const int lane = threadIdx.x & 31;
    const int widx = threadIdx.x >> 5;
    const int wid  = blockIdx.x * 8 + widx;          // 1024 waves
    const int b  = wid >> 8;
    const int s0 = (wid & 255) * 16;                 // 16 query rows per wave
    const int half = lane >> 4, r = lane & 15;

    // Phase A: logits[16 x 64] = Q_tile @ AT^T   (4 N-tiles of 16 agents)
    const float* qb = q + ((size_t)b * SEQ + s0) * DMODEL;
    const unsigned short* atb = at + (size_t)b * AGENTS * DMODEL;
    v8f acc0 = V8F_ZERO, acc1 = V8F_ZERO, acc2 = V8F_ZERO, acc3 = V8F_ZERO;
    for (int k0 = 0; k0 < DMODEL; k0 += 32) {
        __builtin_prefetch(qb + k0 + 32, 0, 0);      // global_prefetch_b8 for next K block
        v16bf af = frag_f32_k(qb + k0, DMODEL, lane);
        acc0 = wmma_bf16(af, frag_bf16_k(atb + (size_t)0  * DMODEL + k0, DMODEL, lane), acc0);
        acc1 = wmma_bf16(af, frag_bf16_k(atb + (size_t)16 * DMODEL + k0, DMODEL, lane), acc1);
        acc2 = wmma_bf16(af, frag_bf16_k(atb + (size_t)32 * DMODEL + k0, DMODEL, lane), acc2);
        acc3 = wmma_bf16(af, frag_bf16_k(atb + (size_t)48 * DMODEL + k0, DMODEL, lane), acc3);
    }

    // Phase B: softmax across the 64 agents of each row.
    // Row m = half*8+j lives in fixed (j,half), spread over 16 lanes (r) x 4 regs.
    float p0[8], p1[8], p2[8], p3[8];
#pragma unroll
    for (int j = 0; j < 8; j++) {
        float l0 = acc0[j] * SCALE, l1 = acc1[j] * SCALE;
        float l2 = acc2[j] * SCALE, l3 = acc3[j] * SCALE;
        float mx = fmaxf(fmaxf(l0, l1), fmaxf(l2, l3));
#pragma unroll
        for (int msk = 1; msk < 16; msk <<= 1) mx = fmaxf(mx, __shfl_xor(mx, msk, 16));
        float e0 = __expf(l0 - mx), e1 = __expf(l1 - mx);
        float e2 = __expf(l2 - mx), e3 = __expf(l3 - mx);
        float sm = e0 + e1 + e2 + e3;
#pragma unroll
        for (int msk = 1; msk < 16; msk <<= 1) sm += __shfl_xor(sm, msk, 16);
        float inv = 1.0f / sm;
        p0[j] = e0 * inv; p1[j] = e1 * inv; p2[j] = e2 * inv; p3[j] = e3 * inv;
    }

    // Phase C: stage a2 (bf16) to LDS in [m][k] order so we can re-load as A fragments.
    unsigned short* chunk = lds_a2 + widx * (16 * 64);
#pragma unroll
    for (int j = 0; j < 8; j++) {
        int row = (half * 8 + j) * 64;
        chunk[row +  0 + r] = f2bf(p0[j]);
        chunk[row + 16 + r] = f2bf(p1[j]);
        chunk[row + 32 + r] = f2bf(p2[j]);
        chunk[row + 48 + r] = f2bf(p3[j]);
    }
    __syncthreads();

    v16bf a2f[2];
#pragma unroll
    for (int ks = 0; ks < 2; ks++) {
        Frag fr;
        const unsigned short* lp = chunk + r * 64 + ks * 32 + half * 8;
        fr.q[0] = *reinterpret_cast<const uint4*>(lp);
        fr.q[1] = *reinterpret_cast<const uint4*>(lp + 16);
        a2f[ks] = fr.bf;
    }

    // Phase D: out_tile[16 x 1024] = a2[16 x 64] @ AV[64 x 1024]  (AV^T rows are K-contig)
    const unsigned short* avb = avT + (size_t)b * DMODEL * AGENTS;
    for (int nt2 = 0; nt2 < 64; nt2++) {
        v8f oc = V8F_ZERO;
        oc = wmma_bf16(a2f[0], frag_bf16_k(avb + (size_t)(nt2 * 16) * AGENTS +  0, AGENTS, lane), oc);
        oc = wmma_bf16(a2f[1], frag_bf16_k(avb + (size_t)(nt2 * 16) * AGENTS + 32, AGENTS, lane), oc);
        int n = nt2 * 16 + r;
        int h = n >> 6, hd = n & 63;
#pragma unroll
        for (int j = 0; j < 8; j++) {
            int m = s0 + half * 8 + j;
            float val = oc[j];
            out[((size_t)b * SEQ + m) * DMODEL + n] = val;
            attn[(((size_t)b * HEADS + h) * SEQ + m) * HDIM + hd] = val;
        }
    }
}

extern "C" void kernel_launch(void* const* d_in, const int* in_sizes, int n_in,
                              void* d_out, int out_size, void* d_ws, size_t ws_size,
                              hipStream_t stream) {
    const float* q    = (const float*)d_in[0];
    const float* k    = (const float*)d_in[1];
    const float* v    = (const float*)d_in[2];
    const float* w    = (const float*)d_in[3];
    const float* bias = (const float*)d_in[4];

    // Workspace layout (~7 MB total)
    char* ws = (char*)d_ws;
    unsigned short* at     = (unsigned short*)(ws);                       // 4*64*1024 bf16  = 512 KB
    float*          logits = (float*)         (ws + 524288);              // 4*64*4096 f32   = 4 MB
    unsigned short* a1     = (unsigned short*)(ws + 524288 + 4194304);    // 4*64*4096 bf16  = 2 MB
    unsigned short* avT    = (unsigned short*)(ws + 524288 + 4194304 + 2097152); // 512 KB

    float* out  = (float*)d_out;
    float* attn = out + (size_t)BATCH * SEQ * DMODEL;

    k_agent_tokens<<<128, 256, 0, stream>>>(q, w, bias, at);
    k_s1_logits  <<<512, 256, 0, stream>>>(at, k, logits);
    k_s1_softmax <<<256, 256, 0, stream>>>(logits, a1);
    k_s1_av      <<<256, 128, 0, stream>>>(a1, v, avT);
    k_s2_fused   <<<128, 256, 0, stream>>>(q, at, avT, out, attn);
}
